// TPExperts_36146444763954
// MI455X (gfx1250) — matrix operand order repaired and checked by reading, not compiled, not run on hardware
//
#include <hip/hip_runtime.h>
#include <hip/hip_bf16.h>
#include <math.h>

// Problem sizes: G=4, E=16, C=2048, H=512, F=2048
#define Gg 4
#define Ee 16
#define Cc 2048
#define Hh 512
#define Ff 2048

// Tiling: 64 tokens per workgroup, F chunked by 64, 16 waves (512 threads)
#define TM 64
#define FC 64
#define NT 512
#define NFC (Ff / FC)

// LDS row strides (bf16 elems), padded for 16B alignment / bank spread
#define XS  (Hh + 8)   // x tile   [TM][XS]
#define WS1 (Hh + 8)   // w1^T     [FC][WS1]  (N-major, K contiguous)
#define WS2 (FC + 8)   // w2^T     [Hh][WS2]  (N-major, K contiguous)
#define AS  (FC + 8)   // gelu act [TM][AS]

typedef __attribute__((ext_vector_type(16))) __bf16 v16bf;
typedef __attribute__((ext_vector_type(4)))  __bf16 v4bf;
typedef __attribute__((ext_vector_type(8)))  float  v8f;

union Frag { v16bf v; uint4 u[2]; };

__device__ __forceinline__ void ld_frag_a(Frag& f, const __bf16* base) {
    // A 16x32 bf16 frag: per lane 16 contiguous K values (32B)
    const uint4* p = reinterpret_cast<const uint4*>(base);
    f.u[0] = p[0];
    f.u[1] = p[1];
}
__device__ __forceinline__ void ld_frag_b(Frag& f, const __bf16* base) {
    // B 32x16 bf16 frag: per lane 16 contiguous K values (32B)
    const uint4* p = reinterpret_cast<const uint4*>(base);
    f.u[0] = p[0];
    f.u[1] = p[1];
}
__device__ __forceinline__ v8f wmma_bf16(const Frag& a, const Frag& b, v8f c) {
    return __builtin_amdgcn_wmma_f32_16x16x32_bf16(false, a.v, false, b.v,
                                                   (short)0, c, false, false);
}

__global__ __launch_bounds__(NT, 1)
void moe_ffn_bf16_wmma(const float* __restrict__ x,
                       const float* __restrict__ w1,
                       const float* __restrict__ b1,
                       const float* __restrict__ w2,
                       const float* __restrict__ b2,
                       float* __restrict__ out)
{
    __shared__ __bf16 s_x[TM * XS];    // 66,560 B
    __shared__ __bf16 s_w[Hh * WS2];   // 73,728 B  (shared W1^T / W2^T staging)
    __shared__ __bf16 s_a[TM * AS];    //  9,216 B

    const int tid  = threadIdx.x;
    const int lane = tid & 31;
    const int wv   = tid >> 5;
    const int half = lane >> 4;
    const int l16  = lane & 15;

    const int e    = blockIdx.y;
    const int tile = blockIdx.x;
    const int tok0 = tile * TM;
    const int g    = tok0 >> 11;           // TM divides C -> tile stays in one g
    const int c0   = tok0 & (Cc - 1);

    const float* xg  = x   + ((size_t)(g * Ee + e) * Cc + c0) * Hh;
    const float* w1g = w1  + (size_t)e * Hh * Ff;
    const float* w2g = w2  + (size_t)e * Ff * Hh;
    const float* b1g = b1  + (size_t)e * Ff;
    const float* b2g = b2  + (size_t)e * Hh;
    float*       og  = out + ((size_t)(g * Ee + e) * Cc + c0) * Hh;

    // ---- single register-staging buffer for BOTH weight chunks ----
    // Live ranges are disjoint (W2 chunk: stage-1 window; next W1 chunk: stage-2
    // window), so one array halves peak VGPR pressure vs. two.
    float rw[16][4];

    auto load_w1 = [&](int f0) {            // rw[i][j] = w1[k4+j][f]
        #pragma unroll
        for (int i = 0; i < 16; ++i) {
            const int qi = i * NT + tid;
            const int fl = qi & (FC - 1);
            const int k4 = (qi >> 6) * 4;
            #pragma unroll
            for (int j = 0; j < 4; ++j)
                rw[i][j] = w1g[(size_t)(k4 + j) * Ff + f0 + fl];    // coalesced b32
        }
    };
    auto store_w1 = [&]() {                 // s_w as W1^T [FC][WS1]
        #pragma unroll
        for (int i = 0; i < 16; ++i) {
            const int qi = i * NT + tid;
            const int fl = qi & (FC - 1);
            const int k4 = (qi >> 6) * 4;
            v4bf bv;
            bv[0] = (__bf16)rw[i][0]; bv[1] = (__bf16)rw[i][1];
            bv[2] = (__bf16)rw[i][2]; bv[3] = (__bf16)rw[i][3];
            *reinterpret_cast<v4bf*>(&s_w[fl * WS1 + k4]) = bv;     // ds_store_b64
        }
    };
    auto load_w2 = [&](int f0) {            // rw[i][j] = w2[fl4+j][h]
        #pragma unroll
        for (int i = 0; i < 16; ++i) {
            const int qi  = i * NT + tid;
            const int h   = qi & (Hh - 1);
            const int fl4 = (qi >> 9) * 4;
            #pragma unroll
            for (int j = 0; j < 4; ++j)
                rw[i][j] = w2g[(size_t)(f0 + fl4 + j) * Hh + h];    // coalesced b32
        }
    };
    auto store_w2 = [&]() {                 // s_w as W2^T [Hh][WS2]
        #pragma unroll
        for (int i = 0; i < 16; ++i) {
            const int qi  = i * NT + tid;
            const int h   = qi & (Hh - 1);
            const int fl4 = (qi >> 9) * 4;
            v4bf bv;
            bv[0] = (__bf16)rw[i][0]; bv[1] = (__bf16)rw[i][1];
            bv[2] = (__bf16)rw[i][2]; bv[3] = (__bf16)rw[i][3];
            *reinterpret_cast<v4bf*>(&s_w[h * WS2 + fl4]) = bv;     // ds_store_b64
        }
    };

    // ---- x tile [TM][H] fp32 -> bf16 LDS, row-major ----
    #pragma unroll
    for (int i = 0; i < (TM * Hh) / (4 * NT); ++i) {
        const int idx = (i * NT + tid) * 4;
        const int row = idx >> 9;
        const int col = idx & (Hh - 1);
        const float4 f = *reinterpret_cast<const float4*>(xg + (size_t)row * Hh + col);
        v4bf bv;
        bv[0] = (__bf16)f.x; bv[1] = (__bf16)f.y; bv[2] = (__bf16)f.z; bv[3] = (__bf16)f.w;
        *reinterpret_cast<v4bf*>(&s_x[row * XS + col]) = bv;
    }

    // Wave tile coordinates (4x4 wave grid, both stages)
    const int wm = wv & 3;
    const int wn = wv >> 2;
    const int m0 = wm * 16;
    const int n0 = wn * 16;      // stage-1 col tile
    const int N0 = wn * 128;     // stage-2 col strip

    // Persistent h2 accumulators: [16 rows x 128 cols] = 8 tiles
    v8f c2[8];
    #pragma unroll
    for (int t = 0; t < 8; ++t)
        #pragma unroll
        for (int r = 0; r < 8; ++r) c2[t][r] = 0.0f;

    load_w1(0);   // prologue: W1 chunk 0 in flight while x tile stages

    for (int fc = 0; fc < NFC; ++fc) {
        const int f0 = fc * FC;

        __syncthreads();                 // prev stage-2 done reading s_w
        store_w1();                      // cvt + b64 stores (W1^T), frees rw
        load_w2(f0);                     // W2 chunk in flight during stage 1
        __syncthreads();                 // W1^T ready

        const float bias = b1g[f0 + n0 + l16];   // issue early

        // ---- stage 1: c1 = x[m0:,:] @ W1[:, f0+n0 : +16], K = 512 ----
        v8f c1;
        #pragma unroll
        for (int r = 0; r < 8; ++r) c1[r] = 0.0f;
        {
            const __bf16* aRow = &s_x[(m0 + l16) * XS + half * 16];
            const __bf16* bRow = &s_w[(n0 + l16) * WS1 + half * 16];
            Frag A[2], B[2];
            ld_frag_a(A[0], aRow);
            ld_frag_b(B[0], bRow);
            #pragma unroll
            for (int ks = 0; ks < Hh / 32; ++ks) {
                const int cur = ks & 1, nxt = cur ^ 1;
                if (ks < Hh / 32 - 1) {              // prefetch next frags
                    ld_frag_a(A[nxt], aRow + (ks + 1) * 32);
                    ld_frag_b(B[nxt], bRow + (ks + 1) * 32);
                }
                c1 = wmma_bf16(A[cur], B[cur], c1);
            }
        }

        // ---- bias + exact GELU -> bf16 activations in LDS ----
        #pragma unroll
        for (int r = 0; r < 8; ++r) {
            const float v  = c1[r] + bias;
            const float gl = 0.5f * v * (1.0f + erff(v * 0.70710678118654752f));
            s_a[(m0 + r + half * 8) * AS + n0 + l16] = (__bf16)gl;
        }

        __syncthreads();                 // stage-1 reads of s_w done, s_a written
        store_w2();                      // cvt + b64 stores (W2^T), frees rw
        if (fc + 1 < NFC) load_w1(f0 + FC);  // next W1 chunk during stage 2
        __syncthreads();                 // W2^T + s_a ready

        // ---- stage 2: c2 += a[m0:,:] @ W2[f0:f0+FC, N0:N0+128], K = FC ----
        {
            const __bf16* aRow = &s_a[(m0 + l16) * AS + half * 16];
            const __bf16* bCol = &s_w[l16 * WS2 + half * 16];
            Frag A2[2], B2[2];
            ld_frag_a(A2[0], aRow);
            ld_frag_b(B2[0], bCol + (size_t)N0 * WS2);
            #pragma unroll
            for (int ks = 0; ks < FC / 32; ++ks) {
                #pragma unroll
                for (int t = 0; t < 8; ++t) {
                    const int i   = ks * 8 + t;
                    const int cur = i & 1, nxt = cur ^ 1;
                    if (i < 15) {                    // prefetch next B frag (+ A on ks flip)
                        const int tn  = (t + 1) & 7;
                        const int ksn = (t == 7) ? ks + 1 : ks;
                        ld_frag_b(B2[nxt], bCol + (size_t)(N0 + tn * 16) * WS2 + ksn * 32);
                        if (t == 7) ld_frag_a(A2[1], aRow + 32);
                    }
                    c2[t] = wmma_bf16(A2[ks], B2[cur], c2[t]);
                }
            }
        }
    }

    // ---- epilogue: + b2, store fp32 ----
    #pragma unroll
    for (int t = 0; t < 8; ++t) {
        const int col = N0 + t * 16 + l16;
        const float bb = b2g[col];
        #pragma unroll
        for (int r = 0; r < 8; ++r) {
            const int row = m0 + r + half * 8;
            og[(size_t)row * Hh + col] = c2[t][r] + bb;
        }
    }
}

extern "C" void kernel_launch(void* const* d_in, const int* in_sizes, int n_in,
                              void* d_out, int out_size, void* d_ws, size_t ws_size,
                              hipStream_t stream) {
    (void)in_sizes; (void)n_in; (void)d_ws; (void)ws_size; (void)out_size;
    const float* x  = (const float*)d_in[0];
    const float* w1 = (const float*)d_in[1];
    const float* b1 = (const float*)d_in[2];
    const float* w2 = (const float*)d_in[3];
    const float* b2 = (const float*)d_in[4];
    float* out = (float*)d_out;

    dim3 grid((Gg * Cc) / TM, Ee, 1);
    dim3 block(NT, 1, 1);
    moe_ffn_bf16_wmma<<<grid, block, 0, stream>>>(x, w1, b1, w2, b2, out);
}
